// Qwen3Attention_26740466385708
// MI455X (gfx1250) — matrix operand order repaired and checked by reading, not compiled, not run on hardware
//
#include <hip/hip_runtime.h>
#include <hip/hip_bf16.h>

// ---------------------------------------------------------------------------
// Qwen3 attention block for MI455X (gfx1250): bf16 WMMA GEMMs + flash attention
// with double-buffered async global->LDS staging (ASYNCcnt software pipeline).
// ---------------------------------------------------------------------------

#define B_BATCH 2
#define S_LEN   2048
#define HIDDEN  1024
#define HEADS   16
#define KV_HEADS 8
#define HEAD_DIM 128
#define RANK    16
#define LORA_SCALE 2.0f
#define RMS_EPS 1e-6f
#define SM_SCALE 0.08838834764831845f   // 128^-0.5
#define MROWS   (B_BATCH * S_LEN)       // 4096

typedef __attribute__((ext_vector_type(16))) __bf16 v16bf;
typedef __attribute__((ext_vector_type(8)))  float  v8f;

__device__ __forceinline__ v8f wmma_bf16(v16bf a, v16bf b, v8f c) {
    // D = A(16x32 bf16) * B(32x16 bf16) + C(16x16 f32)
    return __builtin_amdgcn_wmma_f32_16x16x32_bf16(false, a, false, b,
                                                   (short)0, c, false, false);
}

// Async DMA: 16 bytes per lane, global -> LDS, tracked with ASYNCcnt.
// LDS address = low 32 bits of the generic pointer (ISA 10.2).
__device__ __forceinline__ void async_copy_b128(void* lds_dst, const void* gsrc) {
    unsigned l = (unsigned)(size_t)lds_dst;
    asm volatile("global_load_async_to_lds_b128 %0, %1, off"
                 :: "v"(l), "v"(gsrc) : "memory");
}
__device__ __forceinline__ void wait_async0(void) {
    asm volatile("s_wait_asynccnt 0" ::: "memory");
}
// Leave the 3 in-flight next-tile copies outstanding (async loads complete
// in order, so asynccnt<=3 proves the previous 3 finished).
__device__ __forceinline__ void wait_async3(void) {
    asm volatile("s_wait_asynccnt 3" ::: "memory");
}

// ---------------------------------------------------------------------------
// fp32 -> bf16 cast
// ---------------------------------------------------------------------------
__global__ void cast_f32_bf16(const float* __restrict__ src,
                              __bf16* __restrict__ dst, int n) {
    int i = (blockIdx.x * blockDim.x + threadIdx.x) * 4;
    if (i + 3 < n) {
        float4 v = *(const float4*)(src + i);
        dst[i + 0] = (__bf16)v.x;
        dst[i + 1] = (__bf16)v.y;
        dst[i + 2] = (__bf16)v.z;
        dst[i + 3] = (__bf16)v.w;
    } else {
        for (int j = i; j < n; ++j) dst[j] = (__bf16)src[j];
    }
}

// ---------------------------------------------------------------------------
// LoRA mid: T[M][16] = X[M][K] @ A[16][K]^T  (bf16 out)
// ---------------------------------------------------------------------------
__global__ void lora_mid(const __bf16* __restrict__ X,
                         const float* __restrict__ A,
                         __bf16* __restrict__ T, int M, int K) {
    int idx = blockIdx.x * blockDim.x + threadIdx.x;
    if (idx >= M * RANK) return;
    int m = idx >> 4, j = idx & 15;
    const __bf16* xr = X + (size_t)m * K;
    const float*  ar = A + (size_t)j * K;
    float s = 0.f;
    for (int k = 0; k < K; ++k) s += (float)xr[k] * ar[k];
    T[idx] = (__bf16)s;
}

// ---------------------------------------------------------------------------
// Tiled WMMA GEMM with LoRA epilogue:
//   C[M][N](f32) = X[M][K](bf16) @ W[N][K]^T + scale * T[M][16] @ Bl[N][16]^T
// Block: 256 thr (8 waves), tile 64(M) x 128(N), K step 32.
// Double-buffered async global->LDS pipeline: tile i+1's DMA overlaps tile i's
// WMMAs; s_wait_asynccnt 3 retires only the current tile's copies.
// ---------------------------------------------------------------------------
__global__ __launch_bounds__(256)
void gemm_lora(const __bf16* __restrict__ X, const __bf16* __restrict__ W,
               const __bf16* __restrict__ T, const __bf16* __restrict__ Bl,
               float* __restrict__ C, int M, int N, int K, float scale) {
    __shared__ __align__(16) __bf16 Xs[2][64][40];
    __shared__ __align__(16) __bf16 Ws[2][128][40];

    const int tid  = threadIdx.x;
    const int wid  = tid >> 5;
    const int lane = tid & 31;
    const int lg   = lane & 15;
    const bool hi  = lane >= 16;

    const int mbase = blockIdx.y * 64;
    const int nbase = blockIdx.x * 128;
    const int m0 = (wid >> 2) * 32;   // 0 or 32
    const int n0 = (wid & 3)  * 32;   // 0..96

    v8f acc[2][2];
    for (int i = 0; i < 2; ++i)
        for (int j = 0; j < 2; ++j)
            for (int r = 0; r < 8; ++r) acc[i][j][r] = 0.f;

    const int xr = tid >> 2, xcg = tid & 3;   // X: 64 rows x 32 cols
    const int wr = tid >> 1, whg = tid & 1;   // W: 128 rows x 32 cols

    // issue 3 async b128 copies for k-chunk k0 into buffer `buf`
    auto stage = [&](int k0, int buf) {
        async_copy_b128(&Xs[buf][xr][xcg * 8],
                        X + (size_t)(mbase + xr) * K + k0 + xcg * 8);
        async_copy_b128(&Ws[buf][wr][whg * 16],
                        W + (size_t)(nbase + wr) * K + k0 + whg * 16);
        async_copy_b128(&Ws[buf][wr][whg * 16 + 8],
                        W + (size_t)(nbase + wr) * K + k0 + whg * 16 + 8);
    };

    const int nk = K >> 5;   // K / 32
    stage(0, 0);             // prologue: tile 0 in flight

    for (int it = 0; it < nk; ++it) {
        const int cur = it & 1;
        if (it + 1 < nk) {
            stage((it + 1) << 5, cur ^ 1);  // overlap next tile's DMA
            wait_async3();                   // current tile resident
        } else {
            wait_async0();
        }
        __syncthreads();   // all waves' DMA for `cur` visible

        v16bf a0, a1, b0, b1;
#pragma unroll
        for (int e = 0; e < 16; ++e) {
            int ka = (hi ? 8 : 0) + (e < 8 ? e : e + 8);
            a0[e] = Xs[cur][m0 + lg][ka];
            a1[e] = Xs[cur][m0 + 16 + lg][ka];
            int kb = (hi ? 8 : 0) + (e < 8 ? e : e + 8);
            b0[e] = Ws[cur][n0 + lg][kb];
            b1[e] = Ws[cur][n0 + 16 + lg][kb];
        }
        acc[0][0] = wmma_bf16(a0, b0, acc[0][0]);
        acc[0][1] = wmma_bf16(a0, b1, acc[0][1]);
        acc[1][0] = wmma_bf16(a1, b0, acc[1][0]);
        acc[1][1] = wmma_bf16(a1, b1, acc[1][1]);

        __syncthreads();   // everyone done reading `cur` before it is re-filled
    }

    // LoRA rank-16 epilogue + store
    for (int ms = 0; ms < 2; ++ms)
        for (int ns = 0; ns < 2; ++ns) {
            int n = nbase + n0 + ns * 16 + lg;
            float blr[16];
#pragma unroll
            for (int j = 0; j < 16; ++j) blr[j] = (float)Bl[(size_t)n * 16 + j];
#pragma unroll
            for (int r = 0; r < 8; ++r) {
                int m = mbase + m0 + ms * 16 + r + (hi ? 8 : 0);
                float lor = 0.f;
#pragma unroll
                for (int j = 0; j < 16; ++j)
                    lor += (float)T[(size_t)m * 16 + j] * blr[j];
                C[(size_t)m * N + n] = acc[ms][ns][r] + scale * lor;
            }
        }
}

// ---------------------------------------------------------------------------
// RMSNorm + RoPE + transpose to (B, H, S, D) bf16.
// ---------------------------------------------------------------------------
__global__ void norm_rope(const float* __restrict__ proj,
                          const float* __restrict__ cosT,
                          const float* __restrict__ sinT,
                          const float* __restrict__ nw,
                          __bf16* __restrict__ out, int nheads) {
    int s = blockIdx.x, h = blockIdx.y, b = blockIdx.z, d = threadIdx.x;
    int wid = threadIdx.x >> 5, lane = threadIdx.x & 31;
    __shared__ float red[4];
    __shared__ float xs[HEAD_DIM];

    float x = proj[((size_t)(b * S_LEN + s) * nheads + h) * HEAD_DIM + d];
    float v = x * x;
    for (int off = 1; off < 32; off <<= 1) v += __shfl_xor(v, off, 32);
    if (lane == 0) red[wid] = v;
    __syncthreads();
    float tot = red[0] + red[1] + red[2] + red[3];
    float inv = rsqrtf(tot * (1.0f / HEAD_DIM) + RMS_EPS);
    float xn = x * inv * nw[d];
    xs[d] = xn;
    __syncthreads();
    float other = xs[(d + 64) & 127];
    float c  = cosT[(size_t)(b * S_LEN + s) * HEAD_DIM + d];
    float sn = sinT[(size_t)(b * S_LEN + s) * HEAD_DIM + d];
    float rot = (d < 64) ? -other : other;
    out[((size_t)(b * nheads + h) * S_LEN + s) * HEAD_DIM + d] = (__bf16)(xn * c + rot * sn);
}

__global__ void v_reorder(const float* __restrict__ proj, __bf16* __restrict__ out) {
    int s = blockIdx.x, h = blockIdx.y, b = blockIdx.z, d = threadIdx.x;
    float x = proj[((size_t)(b * S_LEN + s) * KV_HEADS + h) * HEAD_DIM + d];
    out[((size_t)(b * KV_HEADS + h) * S_LEN + s) * HEAD_DIM + d] = (__bf16)x;
}

// ---------------------------------------------------------------------------
// Flash attention: block = 8 waves * 16 q-rows = 128 queries.
// Grid (S/128, H, B). Causal, GQA (q head h -> kv head h/2).
// K staged via async DMA (row-major); V staged transposed (d-major) so the
// B fragments of P*V vectorize to ds_load_b128.
// ---------------------------------------------------------------------------
__global__ __launch_bounds__(256)
void flash_attn(const __bf16* __restrict__ Q, const __bf16* __restrict__ K,
                const __bf16* __restrict__ V, __bf16* __restrict__ O) {
    const int qb = blockIdx.x, h = blockIdx.y, b = blockIdx.z;
    const int kvh = h >> 1;
    const int tid = threadIdx.x, wid = tid >> 5, lane = tid & 31;
    const int lg = lane & 15;
    const bool hi = lane >= 16;

    __shared__ __align__(16) __bf16 Ks[32][136];   // [key][d]
    __shared__ __align__(16) __bf16 Vt[128][40];   // [d][key]  (transposed)
    __shared__ __align__(16) __bf16 Pb[8][16][40]; // per-wave P scratch

    const __bf16* Qbase = Q + (size_t)(b * HEADS + h) * S_LEN * HEAD_DIM;
    const __bf16* Kbase = K + (size_t)(b * KV_HEADS + kvh) * S_LEN * HEAD_DIM;
    const __bf16* Vbase = V + (size_t)(b * KV_HEADS + kvh) * S_LEN * HEAD_DIM;

    const int qrow = qb * 128 + wid * 16 + lg;

    // Load Q fragments (16 rows x 128 dims = 4 A fragments of 16x32)
    v16bf qf[4];
#pragma unroll
    for (int c = 0; c < 4; ++c)
#pragma unroll
        for (int e = 0; e < 16; ++e) {
            int d = c * 32 + (hi ? 8 : 0) + (e < 8 ? e : e + 8);
            qf[c][e] = Qbase[(size_t)qrow * HEAD_DIM + d];
        }

    v8f acc[8];
    for (int nc = 0; nc < 8; ++nc)
        for (int r = 0; r < 8; ++r) acc[nc][r] = 0.f;
    float mrow[8], lrow[8];
    for (int r = 0; r < 8; ++r) { mrow[r] = -1e30f; lrow[r] = 0.f; }

    const int srow = tid >> 3, scg = tid & 7;   // staging: 32 keys x 128 dims

    const int nkt = (qb + 1) * 4;   // 32-key tiles covering [0, qb*128+128)
    for (int kt = 0; kt < nkt; ++kt) {
        const int kb = kt * 32;
        __syncthreads();   // previous tile fully consumed
        // prefetch next tile while we work on this one
        if (kt + 1 < nkt) {
            __builtin_prefetch(Kbase + (size_t)(kb + 32 + srow) * HEAD_DIM + scg * 16, 0, 1);
            __builtin_prefetch(Vbase + (size_t)(kb + 32 + srow) * HEAD_DIM + scg * 16, 0, 1);
        }
        // K: async DMA straight into LDS (row-major)
        async_copy_b128(&Ks[srow][scg * 16],
                        Kbase + (size_t)(kb + srow) * HEAD_DIM + scg * 16);
        async_copy_b128(&Ks[srow][scg * 16 + 8],
                        Kbase + (size_t)(kb + srow) * HEAD_DIM + scg * 16 + 8);
        // V: load 16 halves and scatter transposed into Vt[d][key]
        {
            const uint4* vp = (const uint4*)(Vbase + (size_t)(kb + srow) * HEAD_DIM + scg * 16);
            union { uint4 u[2]; __bf16 hh[16]; } tv;
            tv.u[0] = vp[0];
            tv.u[1] = vp[1];
#pragma unroll
            for (int j = 0; j < 16; ++j) Vt[scg * 16 + j][srow] = tv.hh[j];
        }
        wait_async0();
        __syncthreads();

        // S = Q K^T for keys [kb, kb+32): two 16x16 score tiles
        v8f s0, s1;
        for (int r = 0; r < 8; ++r) { s0[r] = 0.f; s1[r] = 0.f; }
#pragma unroll
        for (int c = 0; c < 4; ++c) {
            v16bf bk0, bk1;
#pragma unroll
            for (int e = 0; e < 16; ++e) {
                int d = c * 32 + (hi ? 8 : 0) + (e < 8 ? e : e + 8);
                bk0[e] = Ks[lg][d];        // key = kb + lg
                bk1[e] = Ks[16 + lg][d];   // key = kb + 16 + lg
            }
            s0 = wmma_bf16(qf[c], bk0, s0);
            s1 = wmma_bf16(qf[c], bk1, s1);
        }

        // online softmax (rows live in 16-lane groups matching C layout)
#pragma unroll
        for (int r = 0; r < 8; ++r) {
            int qi  = qb * 128 + wid * 16 + r + (hi ? 8 : 0);
            int k0i = kb + lg, k1i = kb + 16 + lg;
            float v0 = (k0i <= qi) ? s0[r] * SM_SCALE : -1e30f;
            float v1 = (k1i <= qi) ? s1[r] * SM_SCALE : -1e30f;
            float mv = fmaxf(v0, v1);
            for (int off = 1; off < 16; off <<= 1)
                mv = fmaxf(mv, __shfl_xor(mv, off, 32));
            float mnew  = fmaxf(mrow[r], mv);
            float alpha = __expf(mrow[r] - mnew);
            float p0 = __expf(v0 - mnew);
            float p1 = __expf(v1 - mnew);
            float rs = p0 + p1;
            for (int off = 1; off < 16; off <<= 1)
                rs += __shfl_xor(rs, off, 32);
            lrow[r] = lrow[r] * alpha + rs;
            mrow[r] = mnew;
#pragma unroll
            for (int nc = 0; nc < 8; ++nc) acc[nc][r] *= alpha;
            int mr = r + (hi ? 8 : 0);
            Pb[wid][mr][lg]      = (__bf16)p0;
            Pb[wid][mr][16 + lg] = (__bf16)p1;
        }

        // same-wave LDS RAW: DS ops are in order, just wait for them
        asm volatile("s_wait_dscnt 0" ::: "memory");

        // P (16x32) as A fragment, accumulate P*V over 8 d-chunks
        v16bf pa;
#pragma unroll
        for (int e = 0; e < 16; ++e) {
            int kk = (hi ? 16 : 0) + e;
            pa[e] = Pb[wid][lg][kk];
        }
#pragma unroll
        for (int nc = 0; nc < 8; ++nc) {
            v16bf bv;
#pragma unroll
            for (int e = 0; e < 16; ++e) {
                int kl = (hi ? 16 : 0) + e;
                bv[e] = Vt[nc * 16 + lg][kl];   // contiguous in e -> ds_load_b128
            }
            acc[nc] = wmma_bf16(pa, bv, acc[nc]);
        }
    }

    // normalize and store O: (B, S, H*D) bf16
#pragma unroll
    for (int r = 0; r < 8; ++r) {
        float rl = 1.0f / lrow[r];
        int qi = qb * 128 + wid * 16 + r + (hi ? 8 : 0);
#pragma unroll
        for (int nc = 0; nc < 8; ++nc) {
            int n = h * HEAD_DIM + nc * 16 + lg;
            O[(size_t)(b * S_LEN + qi) * (HEADS * HEAD_DIM) + n] = (__bf16)(acc[nc][r] * rl);
        }
    }
}

// ---------------------------------------------------------------------------
// Launcher
// ---------------------------------------------------------------------------
extern "C" void kernel_launch(void* const* d_in, const int* in_sizes, int n_in,
                              void* d_out, int out_size, void* d_ws, size_t ws_size,
                              hipStream_t stream) {
    const float* hidden = (const float*)d_in[0];
    const float* cosT   = (const float*)d_in[1];
    const float* sinT   = (const float*)d_in[2];
    // d_in[3]: attention_mask (causal) -- recomputed analytically
    const float* q_w = (const float*)d_in[4];
    const float* q_A = (const float*)d_in[5];
    const float* q_B = (const float*)d_in[6];
    const float* k_w = (const float*)d_in[7];
    const float* k_A = (const float*)d_in[8];
    const float* k_B = (const float*)d_in[9];
    const float* v_w = (const float*)d_in[10];
    const float* v_A = (const float*)d_in[11];
    const float* v_B = (const float*)d_in[12];
    const float* o_w = (const float*)d_in[13];
    const float* o_A = (const float*)d_in[14];
    const float* o_B = (const float*)d_in[15];
    const float* q_norm = (const float*)d_in[16];
    const float* k_norm = (const float*)d_in[17];

    size_t off = 0;
    auto alloc = [&](size_t bytes) -> void* {
        void* p = (char*)d_ws + off;
        off += (bytes + 255) & ~(size_t)255;
        return p;
    };
    __bf16* hid_bf  = (__bf16*)alloc((size_t)MROWS * HIDDEN * 2);
    __bf16* qw_bf   = (__bf16*)alloc((size_t)2048 * 1024 * 2);
    __bf16* kw_bf   = (__bf16*)alloc((size_t)1024 * 1024 * 2);
    __bf16* vw_bf   = (__bf16*)alloc((size_t)1024 * 1024 * 2);
    __bf16* ow_bf   = (__bf16*)alloc((size_t)1024 * 2048 * 2);
    __bf16* qB_bf   = (__bf16*)alloc((size_t)2048 * 16 * 2);
    __bf16* kB_bf   = (__bf16*)alloc((size_t)1024 * 16 * 2);
    __bf16* vB_bf   = (__bf16*)alloc((size_t)1024 * 16 * 2);
    __bf16* oB_bf   = (__bf16*)alloc((size_t)1024 * 16 * 2);
    __bf16* tmid    = (__bf16*)alloc((size_t)MROWS * 16 * 2);
    float*  proj    = (float*) alloc((size_t)MROWS * 2048 * 4);  // reused q/k/v
    __bf16* Qbf     = (__bf16*)alloc((size_t)B_BATCH * HEADS * S_LEN * HEAD_DIM * 2);
    __bf16* Kbf     = (__bf16*)alloc((size_t)B_BATCH * KV_HEADS * S_LEN * HEAD_DIM * 2);
    __bf16* Vbf     = (__bf16*)alloc((size_t)B_BATCH * KV_HEADS * S_LEN * HEAD_DIM * 2);
    __bf16* attn_bf = (__bf16*)alloc((size_t)MROWS * 2048 * 2);

    auto cast = [&](const float* s, __bf16* d, int n) {
        int thr = (n + 3) / 4;
        cast_f32_bf16<<<(thr + 255) / 256, 256, 0, stream>>>(s, d, n);
    };
    cast(hidden, hid_bf, MROWS * HIDDEN);
    cast(q_w, qw_bf, 2048 * 1024);
    cast(k_w, kw_bf, 1024 * 1024);
    cast(v_w, vw_bf, 1024 * 1024);
    cast(o_w, ow_bf, 1024 * 2048);
    cast(q_B, qB_bf, 2048 * 16);
    cast(k_B, kB_bf, 1024 * 16);
    cast(v_B, vB_bf, 1024 * 16);
    cast(o_B, oB_bf, 1024 * 16);

    const int midBlocks = (MROWS * 16 + 255) / 256;

    // ---- Q ----
    lora_mid<<<midBlocks, 256, 0, stream>>>(hid_bf, q_A, tmid, MROWS, HIDDEN);
    gemm_lora<<<dim3(2048 / 128, MROWS / 64), 256, 0, stream>>>(
        hid_bf, qw_bf, tmid, qB_bf, proj, MROWS, 2048, HIDDEN, LORA_SCALE);
    norm_rope<<<dim3(S_LEN, HEADS, B_BATCH), 128, 0, stream>>>(
        proj, cosT, sinT, q_norm, Qbf, HEADS);

    // ---- K ----
    lora_mid<<<midBlocks, 256, 0, stream>>>(hid_bf, k_A, tmid, MROWS, HIDDEN);
    gemm_lora<<<dim3(1024 / 128, MROWS / 64), 256, 0, stream>>>(
        hid_bf, kw_bf, tmid, kB_bf, proj, MROWS, 1024, HIDDEN, LORA_SCALE);
    norm_rope<<<dim3(S_LEN, KV_HEADS, B_BATCH), 128, 0, stream>>>(
        proj, cosT, sinT, k_norm, Kbf, KV_HEADS);

    // ---- V ----
    lora_mid<<<midBlocks, 256, 0, stream>>>(hid_bf, v_A, tmid, MROWS, HIDDEN);
    gemm_lora<<<dim3(1024 / 128, MROWS / 64), 256, 0, stream>>>(
        hid_bf, vw_bf, tmid, vB_bf, proj, MROWS, 1024, HIDDEN, LORA_SCALE);
    v_reorder<<<dim3(S_LEN, KV_HEADS, B_BATCH), 128, 0, stream>>>(proj, Vbf);

    // ---- attention ----
    flash_attn<<<dim3(S_LEN / 128, HEADS, B_BATCH), 256, 0, stream>>>(
        Qbf, Kbf, Vbf, attn_bf);

    // ---- output projection ----
    lora_mid<<<midBlocks, 256, 0, stream>>>(attn_bf, o_A, tmid, MROWS, 2048);
    gemm_lora<<<dim3(1024 / 128, MROWS / 64), 256, 0, stream>>>(
        attn_bf, ow_bf, tmid, oB_bf, (float*)d_out, MROWS, 1024, 2048, LORA_SCALE);

    (void)in_sizes; (void)n_in; (void)out_size; (void)ws_size;
}
